// LSTMConv_72816875537074
// MI455X (gfx1250) — matrix operand order repaired and checked by reading, not compiled, not run on hardware
//
#include <hip/hip_runtime.h>
#include <hip/hip_bf16.h>

typedef __attribute__((ext_vector_type(16))) __bf16       v16bf;
typedef __attribute__((ext_vector_type(8)))  float        v8f;
typedef __attribute__((ext_vector_type(4)))  unsigned int u32x4;

#define HIDN    128
#define OUTF    64
#define NSTEP   32
#define C_IN    64
#define P_TOT   16384   // 128*128 patch positions
#define P_TILE  32      // P rows per workgroup
#define KTOT    192     // 64 (input) + 128 (hidden)
#define WSTRIDE 200     // padded bf16 row stride for Wcat and A (bank-conflict-free)
#define FCSTR   136     // padded bf16 row stride for fc_w

// ---- LDS partition (bytes). 235008 B total -> needs CDNA5's 320KB/WG LDS ----
constexpr int LDS_W_OFF  = 0;                               // 512 x WSTRIDE bf16
constexpr int LDS_FC_OFF = 512 * WSTRIDE * 2;               // 64  x FCSTR  bf16
constexpr int LDS_A_OFF  = LDS_FC_OFF + OUTF * FCSTR * 2;   // 32  x WSTRIDE bf16
constexpr int LDS_BYTES  = LDS_A_OFF + P_TILE * WSTRIDE * 2;

// B-operand fragment: lane = col(l%16), K = 16*(l/16) + 0..15 contiguous
__device__ inline v16bf lds_ld_B(const char* p) {
    union { u32x4 q[2]; v16bf v; } u;
    u.q[0] = *(const u32x4*)(p);
    u.q[1] = *(const u32x4*)(p + 16);
    return u.v;
}
// A-operand fragment: lane = row(l%16); K = kbase + 16*(l/16) + 0..15 contiguous
__device__ inline v16bf lds_ld_A(const char* Abase, int row, int kbase, int lanehi) {
    const char* r = Abase + row * (WSTRIDE * 2) + (kbase + lanehi * 16) * 2;
    union { u32x4 q[2]; v16bf v; } u;
    u.q[0] = *(const u32x4*)(r);
    u.q[1] = *(const u32x4*)(r + 16);
    return u.v;
}

__device__ inline float fast_sigmoid(float x) { return 1.0f / (1.0f + __expf(-x)); }
__device__ inline float fast_tanh(float x)    { float e = __expf(2.0f * x); return (e - 1.0f) / (e + 1.0f); }

__global__ __launch_bounds__(256, 1)
void lstm_conv_fused(const float* __restrict__ feats,
                     const float* __restrict__ W_ih,
                     const float* __restrict__ W_hh,
                     const float* __restrict__ b_ih,
                     const float* __restrict__ b_hh,
                     const float* __restrict__ fc_w,
                     const float* __restrict__ fc_b,
                     float* __restrict__ out)
{
    extern __shared__ char smem[];
    char* Wc = smem + LDS_W_OFF;   // [512][WSTRIDE] bf16 : cols 0..63 = W_ih, 64..191 = W_hh
    char* FC = smem + LDS_FC_OFF;  // [64][FCSTR]    bf16
    char* A  = smem + LDS_A_OFF;   // [32][WSTRIDE]  bf16 : cols 0..63 = xs_t, 64..191 = h

    const int tid    = threadIdx.x;
    const int wid    = tid >> 5;          // wave 0..7
    const int lane   = tid & 31;
    const int nl     = lane & 15;
    const int lanehi = lane >> 4;
    const int pbase  = blockIdx.x * P_TILE;

    // ---------- preload weights (f32 global -> bf16 LDS) ----------
    for (int idx = tid; idx < 512 * 8; idx += 256) {            // W_ih
        int n = idx >> 3, kc = idx & 7;
        const float* s = W_ih + n * C_IN + kc * 8;
        union { __bf16 h[8]; u32x4 q; } u;
        #pragma unroll
        for (int j = 0; j < 8; ++j) u.h[j] = (__bf16)s[j];
        *(u32x4*)(Wc + n * (WSTRIDE * 2) + kc * 16) = u.q;
    }
    for (int idx = tid; idx < 512 * 16; idx += 256) {           // W_hh
        int n = idx >> 4, kc = idx & 15;
        const float* s = W_hh + n * HIDN + kc * 8;
        union { __bf16 h[8]; u32x4 q; } u;
        #pragma unroll
        for (int j = 0; j < 8; ++j) u.h[j] = (__bf16)s[j];
        *(u32x4*)(Wc + n * (WSTRIDE * 2) + 128 + kc * 16) = u.q;
    }
    for (int idx = tid; idx < 64 * 16; idx += 256) {            // fc_w
        int o = idx >> 4, kc = idx & 15;
        const float* s = fc_w + o * HIDN + kc * 8;
        union { __bf16 h[8]; u32x4 q; } u;
        #pragma unroll
        for (int j = 0; j < 8; ++j) u.h[j] = (__bf16)s[j];
        *(u32x4*)(FC + o * (FCSTR * 2) + kc * 16) = u.q;
    }
    // h(0) = 0
    {
        u32x4 z = {0u, 0u, 0u, 0u};
        for (int idx = tid; idx < P_TILE * 16; idx += 256) {
            int p = idx >> 4, kc = idx & 15;
            *(u32x4*)(A + p * (WSTRIDE * 2) + 128 + kc * 16) = z;
        }
    }

    // ---------- per-lane constants ----------
    float bq[4];                                   // combined bias for gate columns owned by this lane
    #pragma unroll
    for (int q = 0; q < 4; ++q) {
        int j = q * HIDN + wid * 16 + nl;
        bq[q] = b_ih[j] + b_hh[j];
    }
    const int fmt = wid >> 2, fnt = wid & 3;       // fc GEMM tile: (M-tile, N-tile) per wave
    const float fbias = fc_b[fnt * 16 + nl];

    float cst[2][8];                               // persistent LSTM cell state (in registers)
    #pragma unroll
    for (int mt = 0; mt < 2; ++mt)
        #pragma unroll
        for (int r = 0; r < 8; ++r) cst[mt][r] = 0.0f;

    // xs gather constants: bottom-right pixel of each 2x2 patch
    const int p_ld  = tid & 31;                    // p_local this thread stages
    const int cgrp  = tid >> 5;                    // 8 channels per thread
    const int prow  = 2 * (pbase >> 7) + 1;        // constant image row for whole tile
    const int pcol  = 2 * ((pbase & 127) + p_ld) + 1;

    // ---------- time loop ----------
    for (int n = 0; n < NSTEP; ++n) {
        // stage xs_t -> A[:,0:64] (bf16)
        {
            const float* s = feats + (size_t)(n * C_IN + cgrp * 8) * 65536
                                   + (size_t)prow * 256 + pcol;
            union { __bf16 h[8]; u32x4 q; } u;
            #pragma unroll
            for (int j = 0; j < 8; ++j) u.h[j] = (__bf16)s[(size_t)j * 65536];
            *(u32x4*)(A + p_ld * (WSTRIDE * 2) + cgrp * 16) = u.q;
        }
        __syncthreads();

        // gates = [xs|h] @ [W_ih|W_hh]^T + b   (M=32, N=512, K=192) -> wave-local i/f/g/o tiles
        v8f acc[2][4];
        #pragma unroll
        for (int mt = 0; mt < 2; ++mt)
            #pragma unroll
            for (int q = 0; q < 4; ++q)
                #pragma unroll
                for (int r = 0; r < 8; ++r) acc[mt][q][r] = bq[q];

        #pragma unroll
        for (int ks = 0; ks < 6; ++ks) {
            const int kb = ks * 32;
            v16bf a0 = lds_ld_A(A, nl,      kb, lanehi);
            v16bf a1 = lds_ld_A(A, 16 + nl, kb, lanehi);
            #pragma unroll
            for (int q = 0; q < 4; ++q) {
                const int nrow = q * HIDN + wid * 16 + nl;
                v16bf b = lds_ld_B(Wc + nrow * (WSTRIDE * 2) + kb * 2 + lanehi * 32);
                acc[0][q] = __builtin_amdgcn_wmma_f32_16x16x32_bf16(
                                false, a0, false, b, (short)0, acc[0][q], false, false);
                acc[1][q] = __builtin_amdgcn_wmma_f32_16x16x32_bf16(
                                false, a1, false, b, (short)0, acc[1][q], false, false);
            }
        }
        __syncthreads();   // all waves done reading A before h is overwritten

        // LSTM pointwise in registers; write h (bf16) back to A[:,64:192]
        #pragma unroll
        for (int mt = 0; mt < 2; ++mt) {
            #pragma unroll
            for (int r = 0; r < 8; ++r) {
                float iv = fast_sigmoid(acc[mt][0][r]);
                float fv = fast_sigmoid(acc[mt][1][r]);
                float gv = fast_tanh   (acc[mt][2][r]);
                float ov = fast_sigmoid(acc[mt][3][r]);
                float cv = fv * cst[mt][r] + iv * gv;
                cst[mt][r] = cv;
                float hv = ov * fast_tanh(cv);
                int p = mt * 16 + r + 8 * lanehi;  // C-layout row of this slot
                *(__bf16*)(A + p * (WSTRIDE * 2) + (64 + wid * 16 + nl) * 2) = (__bf16)hv;
            }
        }
        __syncthreads();   // h visible to all waves

        // out_t = h @ fc_w^T + fc_b   (M=32, N=64, K=128); one 16x16 tile per wave
        v8f oacc;
        #pragma unroll
        for (int r = 0; r < 8; ++r) oacc[r] = fbias;
        #pragma unroll
        for (int ks = 0; ks < 4; ++ks) {
            const int kb = ks * 32;
            v16bf a = lds_ld_A(A, fmt * 16 + nl, 64 + kb, lanehi);
            v16bf b = lds_ld_B(FC + (fnt * 16 + nl) * (FCSTR * 2) + kb * 2 + lanehi * 32);
            oacc = __builtin_amdgcn_wmma_f32_16x16x32_bf16(
                       false, a, false, b, (short)0, oacc, false, false);
        }
        float* op = out + (size_t)n * OUTF * P_TOT
                        + (size_t)(fnt * 16 + nl) * P_TOT
                        + pbase + fmt * 16 + 8 * lanehi;
        #pragma unroll
        for (int r = 0; r < 8; ++r) op[r] = oacc[r];
        // next iteration's xs staging touches A cols 0..63 only (disjoint from fc reads)
    }
}

extern "C" void kernel_launch(void* const* d_in, const int* in_sizes, int n_in,
                              void* d_out, int out_size, void* d_ws, size_t ws_size,
                              hipStream_t stream) {
    (void)in_sizes; (void)n_in; (void)out_size; (void)d_ws; (void)ws_size;
    const float* feats = (const float*)d_in[0];
    const float* W_ih  = (const float*)d_in[1];
    const float* W_hh  = (const float*)d_in[2];
    const float* b_ih  = (const float*)d_in[3];
    const float* b_hh  = (const float*)d_in[4];
    const float* fc_w  = (const float*)d_in[5];
    const float* fc_b  = (const float*)d_in[6];

    // 235 KB LDS per workgroup (CDNA5 allows up to 320 KB) — opt in every call (idempotent,
    // immediate API, not a stream op, so it is graph-capture safe and deterministic).
    (void)hipFuncSetAttribute((const void*)lstm_conv_fused,
                              hipFuncAttributeMaxDynamicSharedMemorySize, LDS_BYTES);

    dim3 grid(P_TOT / P_TILE);   // 512 workgroups
    dim3 block(256);             // 8 waves (wave32)
    lstm_conv_fused<<<grid, block, LDS_BYTES, stream>>>(
        feats, W_ih, W_hh, b_ih, b_hh, fc_w, fc_b, (float*)d_out);
}
